// MoELayer_6313601925645
// MI455X (gfx1250) — compile-verified, matrix operand order checked by reading, not run on hardware
//
#include <hip/hip_runtime.h>
#include <hip/hip_fp16.h>
#include <math.h>

// Problem constants (match reference)
#define BB 4
#define NN 2048
#define CC 512
#define FFC 2048
#define EE 8
#define TT (BB * NN)        // 8192 tokens
#define TM 32               // tokens per expert tile
#define CHB 128             // FF columns per fused chunk
#define NCH (FFC / CHB)     // 16 chunks
#define XSTR 520            // padded LDS stride for x tile (halves)
#define HSTR 136            // padded LDS stride for h chunk (halves)

typedef _Float16 f16;
typedef __attribute__((ext_vector_type(16))) _Float16 v16h;
typedef __attribute__((ext_vector_type(8)))  float    v8f;

// ---------------------------------------------------------------------------
// Fragment loaders for V_WMMA_F32_16X16X32_F16 (ISA 7.12.2 16-bit layout):
// lanes 0-15 hold K = {0..7, 16..23}; lanes 16-31 hold K = {8..15, 24..31}.
// f16 source: two contiguous 16B segments -> 2x b128. fp32 fallback: 4x b128+cvt.
// ---------------------------------------------------------------------------
__device__ __forceinline__ v16h frag_from_f16row(const f16* rowp, int lane) {
    const int k1 = (lane & 16) ? 8 : 0;
    v16h r;
#pragma unroll
    for (int i = 0; i < 8; ++i) {
        r[i]     = rowp[k1 + i];
        r[8 + i] = rowp[k1 + 16 + i];
    }
    return r;
}

__device__ __forceinline__ v16h frag_from_f32row(const float* rowp, int lane) {
    const int k1 = (lane & 16) ? 8 : 0;
    const float4 a0 = *(const float4*)(rowp + k1);
    const float4 a1 = *(const float4*)(rowp + k1 + 4);
    const float4 c0 = *(const float4*)(rowp + k1 + 16);
    const float4 c1 = *(const float4*)(rowp + k1 + 20);
    float t[16] = {a0.x, a0.y, a0.z, a0.w, a1.x, a1.y, a1.z, a1.w,
                   c0.x, c0.y, c0.z, c0.w, c1.x, c1.y, c1.z, c1.w};
    v16h r;
#pragma unroll
    for (int i = 0; i < 16; ++i) r[i] = (_Float16)t[i];
    return r;
}

// ---------------------------------------------------------------------------
// Kernel 0: streaming fp32 -> f16 conversion (weights / activations).
// ---------------------------------------------------------------------------
__global__ void cvt_f32_to_f16_kernel(const float* __restrict__ src,
                                      f16* __restrict__ dst, int n8) {
    const int i = blockIdx.x * blockDim.x + threadIdx.x;
    if (i >= n8) return;
    const float4 a = *(const float4*)(src + (size_t)i * 8);
    const float4 b = *(const float4*)(src + (size_t)i * 8 + 4);
    f16* d = dst + (size_t)i * 8;
    d[0] = (_Float16)a.x; d[1] = (_Float16)a.y;
    d[2] = (_Float16)a.z; d[3] = (_Float16)a.w;
    d[4] = (_Float16)b.x; d[5] = (_Float16)b.y;
    d[6] = (_Float16)b.z; d[7] = (_Float16)b.w;
}

// ---------------------------------------------------------------------------
// Kernel 1: router. One thread per token: logits -> softmax -> top-2 ->
// normalized weights. Writes picks (i1 | i2<<8) and two weights per token.
// ---------------------------------------------------------------------------
__global__ void router_kernel(const float* __restrict__ x,
                              const float* __restrict__ router_w,
                              int* __restrict__ picks,
                              float* __restrict__ wts) {
    __shared__ float rw[EE * CC];   // 16 KB
    const int tid = threadIdx.x;
    for (int i = tid; i < EE * CC; i += blockDim.x) rw[i] = router_w[i];
    __syncthreads();

    const int t = blockIdx.x * blockDim.x + tid;
    const float* xr = x + (size_t)t * CC;

    float lg[EE];
#pragma unroll
    for (int e = 0; e < EE; ++e) lg[e] = 0.f;
    for (int c = 0; c < CC; c += 4) {
        const float4 xv = *(const float4*)(xr + c);
#pragma unroll
        for (int e = 0; e < EE; ++e) {
            const float4 wv = *(const float4*)(rw + e * CC + c);
            lg[e] += xv.x * wv.x + xv.y * wv.y + xv.z * wv.z + xv.w * wv.w;
        }
    }
    float m = lg[0];
#pragma unroll
    for (int e = 1; e < EE; ++e) m = fmaxf(m, lg[e]);
    float p[EE], s = 0.f;
#pragma unroll
    for (int e = 0; e < EE; ++e) { p[e] = __expf(lg[e] - m); s += p[e]; }
    const float inv = 1.f / s;
#pragma unroll
    for (int e = 0; e < EE; ++e) p[e] *= inv;
    int i1 = 0;
#pragma unroll
    for (int e = 1; e < EE; ++e) if (p[e] > p[i1]) i1 = e;
    int i2 = (i1 == 0) ? 1 : 0;
#pragma unroll
    for (int e = 0; e < EE; ++e) if (e != i1 && p[e] > p[i2]) i2 = e;
    const float denom = 1.f / (p[i1] + p[i2] + 1e-9f);
    picks[t]       = i1 | (i2 << 8);
    wts[2 * t]     = p[i1] * denom;
    wts[2 * t + 1] = p[i2] * denom;
}

// ---------------------------------------------------------------------------
// Kernel 2: deterministic compaction. One wave32 per expert; order-preserving
// ballot + prefix-popcount compaction of token indices and weights.
// ---------------------------------------------------------------------------
__global__ void compact_kernel(const int* __restrict__ picks,
                               const float* __restrict__ wts,
                               int* __restrict__ cnt,
                               int* __restrict__ idx,
                               float* __restrict__ wgt) {
    const int lane = threadIdx.x & 31;
    const int e    = threadIdx.x >> 5;   // 8 waves, one per expert
    int base = 0;
    for (int t0 = 0; t0 < TT; t0 += 32) {
        const int t = t0 + lane;
        const int p = picks[t];
        const bool s0 = ((p & 0xFF) == e);
        const bool s1 = (((p >> 8) & 0xFF) == e);
        const bool pred = s0 || s1;
        const unsigned mask = (unsigned)__ballot(pred);
        const int off = __popc(mask & ((1u << lane) - 1u));
        if (pred) {
            idx[e * TT + base + off] = t;
            wgt[e * TT + base + off] = s0 ? wts[2 * t] : wts[2 * t + 1];
        }
        base += __popc(mask);
    }
    if (lane == 0) cnt[e] = base;
}

// ---------------------------------------------------------------------------
// Kernel 3: fused expert FFN over gathered tokens.
// Block = 256 threads (8 waves), tile = 32 tokens, expert = blockIdx.y.
// Per FF chunk of 128: h = gelu(x@w1^T + b1) via WMMA, then o += h@w2^T.
// A sched_barrier fences all fragment loads of a group ahead of the WMMAs so
// up to 8 global_load_b128 stay in flight per wait (staggered s_wait_loadcnt).
// WF16: weights/x pre-converted to f16 in workspace; else fp32 + cvt fallback.
// ---------------------------------------------------------------------------
template <bool WF16>
__global__ void __launch_bounds__(256)
expert_fused_kernel(const float* __restrict__ xf,
                    const f16*   __restrict__ xh,
                    const float* __restrict__ w1f,
                    const f16*   __restrict__ w1h,
                    const float* __restrict__ w2f,
                    const f16*   __restrict__ w2h,
                    const float* __restrict__ b1,
                    const float* __restrict__ b2,
                    const int*   __restrict__ cnt,
                    const int*   __restrict__ idx,
                    const float* __restrict__ wgt,
                    float* __restrict__ out) {
    const int e  = blockIdx.y;
    const int m0 = blockIdx.x * TM;
    const int cntE = cnt[e];
    if (m0 >= cntE) return;                       // block-uniform exit
    const int rows = min(TM, cntE - m0);

    __shared__ f16   xs[TM * XSTR];               // ~32.5 KB x tile (f16)
    __shared__ f16   hs[TM * HSTR];               // ~8.5 KB h chunk (f16)
    __shared__ int   tokS[TM];
    __shared__ float cwS[TM];

    const int tid  = threadIdx.x;
    const int lane = tid & 31;
    const int wave = tid >> 5;
    const int r16  = lane & 15;
    const int mhi  = (lane >> 4) * 8;

    if (tid < TM) {
        const bool v = (tid < rows);
        tokS[tid] = v ? idx[e * TT + m0 + tid] : 0;
        cwS[tid]  = v ? wgt[e * TT + m0 + tid] : 0.f;
    }
    __syncthreads();

    // Load x tile (32 x 512) into LDS as f16.
    if (WF16) {
        for (int i = tid; i < TM * (CC / 8); i += 256) {
            const int r = i >> 6;                 // / (CC/8)
            const int c8 = (i & 63) * 8;
            f16 v[8] = {};
            if (r < rows) {
                const uint4 raw = *(const uint4*)(xh + (size_t)tokS[r] * CC + c8);
                *(uint4*)v = raw;
            }
            f16* d = xs + r * XSTR + c8;
#pragma unroll
            for (int q = 0; q < 8; ++q) d[q] = v[q];
        }
    } else {
        for (int i = tid; i < TM * (CC / 4); i += 256) {
            const int r = i >> 7;
            const int c4 = (i & 127) * 4;
            float4 v = make_float4(0.f, 0.f, 0.f, 0.f);
            if (r < rows) v = *(const float4*)(xf + (size_t)tokS[r] * CC + c4);
            f16* d = xs + r * XSTR + c4;
            d[0] = (_Float16)v.x; d[1] = (_Float16)v.y;
            d[2] = (_Float16)v.z; d[3] = (_Float16)v.w;
        }
    }

    const float* w1fe = w1f + (size_t)e * FFC * CC;
    const f16*   w1he = w1h + (size_t)e * FFC * CC;
    const float* w2fe = w2f + (size_t)e * CC * FFC;
    const f16*   w2he = w2h + (size_t)e * CC * FFC;

    v8f acc[8];                                   // [nt][rb] -> acc[nt*2+rb]
#pragma unroll
    for (int n = 0; n < 8; ++n) acc[n] = (v8f)(0.f);

    for (int ch = 0; ch < NCH; ++ch) {
        const int ff0 = ch * CHB;

        // ---- Stage A: wave owns 16 FF cols (fb = wave); both row blocks ----
        const int fcol = ff0 + wave * 16;         // + r16 = B-matrix lane row
        v8f ha0 = (v8f)(0.f), ha1 = (v8f)(0.f);
        const f16* xr0 = xs + r16 * XSTR;
        const f16* xr1 = xs + (16 + r16) * XSTR;
#pragma unroll 2
        for (int kb = 0; kb < CC; kb += 64) {
            // Two B fragments + four A fragments in flight before any wmma.
            v16h b0, b1v_;
            if (WF16) {
                const f16* brow = w1he + (size_t)(fcol + r16) * CC + kb;
                b0   = frag_from_f16row(brow, lane);
                b1v_ = frag_from_f16row(brow + 32, lane);
            } else {
                const float* brow = w1fe + (size_t)(fcol + r16) * CC + kb;
                b0   = frag_from_f32row(brow, lane);
                b1v_ = frag_from_f32row(brow + 32, lane);
            }
            const v16h a00 = frag_from_f16row(xr0 + kb, lane);
            const v16h a10 = frag_from_f16row(xr1 + kb, lane);
            const v16h a01 = frag_from_f16row(xr0 + kb + 32, lane);
            const v16h a11 = frag_from_f16row(xr1 + kb + 32, lane);
            __builtin_amdgcn_sched_barrier(0);   // keep all loads above wmmas
            ha0 = __builtin_amdgcn_wmma_f32_16x16x32_f16(
                false, a00, false, b0, (short)0, ha0, false, false);
            ha1 = __builtin_amdgcn_wmma_f32_16x16x32_f16(
                false, a10, false, b0, (short)0, ha1, false, false);
            ha0 = __builtin_amdgcn_wmma_f32_16x16x32_f16(
                false, a01, false, b1v_, (short)0, ha0, false, false);
            ha1 = __builtin_amdgcn_wmma_f32_16x16x32_f16(
                false, a11, false, b1v_, (short)0, ha1, false, false);
        }
        __syncthreads();   // previous chunk's stage B has consumed hs

        // bias + exact GELU + f16 store to hs (h column = fcol-ff0 + r16)
        const float b1s = b1[e * FFC + fcol + r16];
#pragma unroll
        for (int j = 0; j < 8; ++j) {
            const float h0 = ha0[j] + b1s;
            const float h1 = ha1[j] + b1s;
            const float g0 = 0.5f * h0 * (1.0f + erff(h0 * 0.70710678118f));
            const float g1 = 0.5f * h1 * (1.0f + erff(h1 * 0.70710678118f));
            hs[(j + mhi) * HSTR + wave * 16 + r16]      = (_Float16)g0;
            hs[(16 + j + mhi) * HSTR + wave * 16 + r16] = (_Float16)g1;
        }
        __syncthreads();

        // ---- Stage B: wave owns 64 C cols; o += h[32x128] @ w2_chunk^T ----
        const int c0w = wave * 64;
#pragma unroll
        for (int kb = 0; kb < CHB; kb += 32) {
            // All 4 B fragments (8x global_load_b128) issued before the wmmas.
            v16h barr[4];
#pragma unroll
            for (int nt = 0; nt < 4; ++nt) {
                const int crow = c0w + nt * 16 + r16;
                if (WF16)
                    barr[nt] = frag_from_f16row(w2he + (size_t)crow * FFC + ff0 + kb, lane);
                else
                    barr[nt] = frag_from_f32row(w2fe + (size_t)crow * FFC + ff0 + kb, lane);
            }
            const v16h a0 = frag_from_f16row(hs + r16 * HSTR + kb, lane);
            const v16h a1 = frag_from_f16row(hs + (16 + r16) * HSTR + kb, lane);
            __builtin_amdgcn_sched_barrier(0);   // keep all loads above wmmas
#pragma unroll
            for (int nt = 0; nt < 4; ++nt) {
                acc[nt * 2 + 0] = __builtin_amdgcn_wmma_f32_16x16x32_f16(
                    false, a0, false, barr[nt], (short)0, acc[nt * 2 + 0], false, false);
                acc[nt * 2 + 1] = __builtin_amdgcn_wmma_f32_16x16x32_f16(
                    false, a1, false, barr[nt], (short)0, acc[nt * 2 + 1], false, false);
            }
        }
    }

    // ---- Epilogue: out[token] += cw * (o + b2) (each token in 2 lists) ----
    const int c0w = wave * 64;
#pragma unroll
    for (int nt = 0; nt < 4; ++nt) {
        const int col = c0w + nt * 16 + r16;
        const float b2v = b2[e * CC + col];
#pragma unroll
        for (int rb = 0; rb < 2; ++rb) {
#pragma unroll
            for (int j = 0; j < 8; ++j) {
                const int m = rb * 16 + j + mhi;
                if (m < rows) {
                    atomicAdd(out + (size_t)tokS[m] * CC + col,
                              cwS[m] * (acc[nt * 2 + rb][j] + b2v));
                }
            }
        }
    }
}

// ---------------------------------------------------------------------------
// Host launcher (graph-capture safe: only async ops on `stream`).
// Workspace: picks[T] | wts[2T] | cnt(pad 256B) | idx[E*T] | wgt[E*T] |
//            xh[T*C] f16 | w1h[E*FF*C] f16 | w2h[E*C*FF] f16   (~40.6 MB)
// Falls back to fp32-weight path (on-the-fly cvt) if ws_size is too small.
// ---------------------------------------------------------------------------
extern "C" void kernel_launch(void* const* d_in, const int* in_sizes, int n_in,
                              void* d_out, int out_size, void* d_ws, size_t ws_size,
                              hipStream_t stream) {
    const float* x        = (const float*)d_in[0];
    const float* router_w = (const float*)d_in[1];
    const float* w1       = (const float*)d_in[2];
    const float* b1       = (const float*)d_in[3];
    const float* w2       = (const float*)d_in[4];
    const float* b2       = (const float*)d_in[5];
    float* out = (float*)d_out;

    char* ws = (char*)d_ws;
    size_t off = 0;
    int*   picks = (int*)(ws + off);   off += (size_t)TT * 4;
    float* wts   = (float*)(ws + off); off += (size_t)2 * TT * 4;
    int*   cnt   = (int*)(ws + off);   off += 256;
    int*   idx   = (int*)(ws + off);   off += (size_t)EE * TT * 4;
    float* wgt   = (float*)(ws + off); off += (size_t)EE * TT * 4;
    f16*   xh    = (f16*)(ws + off);   off += (size_t)TT * CC * 2;
    f16*   w1h   = (f16*)(ws + off);   off += (size_t)EE * FFC * CC * 2;
    f16*   w2h   = (f16*)(ws + off);   off += (size_t)EE * CC * FFC * 2;
    const bool f16path = (ws_size >= off);

    hipMemsetAsync(d_out, 0, (size_t)out_size * sizeof(float), stream);

    router_kernel<<<TT / 256, 256, 0, stream>>>(x, router_w, picks, wts);
    compact_kernel<<<1, 256, 0, stream>>>(picks, wts, cnt, idx, wgt);

    if (f16path) {
        const int nx  = TT * CC / 8;
        const int nw  = EE * FFC * CC / 8;
        cvt_f32_to_f16_kernel<<<(nx + 255) / 256, 256, 0, stream>>>(x, xh, nx);
        cvt_f32_to_f16_kernel<<<(nw + 255) / 256, 256, 0, stream>>>(w1, w1h, nw);
        cvt_f32_to_f16_kernel<<<(nw + 255) / 256, 256, 0, stream>>>(w2, w2h, nw);
        expert_fused_kernel<true><<<dim3(TT / TM, EE), 256, 0, stream>>>(
            x, xh, w1, w1h, w2, w2h, b1, b2, cnt, idx, wgt, out);
    } else {
        expert_fused_kernel<false><<<dim3(TT / TM, EE), 256, 0, stream>>>(
            x, xh, w1, w1h, w2, w2h, b1, b2, cnt, idx, wgt, out);
    }
}
